// CausalSTDiT2_7653631722024
// MI455X (gfx1250) — matrix operand order repaired and checked by reading, not compiled, not run on hardware
//
#include <hip/hip_runtime.h>
#include <math.h>

typedef __bf16 bf16_t;
typedef __attribute__((ext_vector_type(16))) __bf16 v16b;
typedef __attribute__((ext_vector_type(8)))  float  v8f;
typedef int v4i_ __attribute__((__vector_size__(16)));

#define B_    2
#define T_    32
#define S_    256
#define C_    1152
#define NH_   16
#define HD_   72
#define N_    (T_*S_)        // 8192 tokens per batch
#define M_    (B_*N_)        // 16384 total tokens
#define C3_   (3*C_)         // 3456
#define LY_   120
#define MLPH_ (4*C_)         // 4608
#define HD2_  36
#define SCALE_ 0.11785113019775793f   // 72^-0.5

static __device__ __forceinline__ bf16_t tobf(float f) { return (bf16_t)f; }
static __device__ __forceinline__ float  fromb(bf16_t b) { return (float)b; }

// ---------------- async global->LDS copy (CDNA5 ASYNCcnt path) ----------------
#if defined(__gfx1250__) && __has_builtin(__builtin_amdgcn_global_load_async_to_lds_b128) && __has_builtin(__builtin_amdgcn_s_wait_asynccnt)
#define ASYNC_OK 1
#else
#define ASYNC_OK 0
#endif

// copy 16 bytes (8 bf16) global -> LDS.  16B-aligned on both sides by construction.
static __device__ __forceinline__ void cp16(const bf16_t* g, bf16_t* l) {
#if ASYNC_OK
  __builtin_amdgcn_global_load_async_to_lds_b128(
      (__attribute__((address_space(1))) v4i_*)(unsigned long long)(__SIZE_TYPE__)g,
      (__attribute__((address_space(3))) v4i_*)(unsigned)(__SIZE_TYPE__)l,
      0, 0);
#else
  *(int4*)l = *(const int4*)g;
#endif
}
static __device__ __forceinline__ void async_wait() {
#if ASYNC_OK
  __builtin_amdgcn_s_wait_asynccnt(0);
#endif
}
static __device__ __forceinline__ void zero16(bf16_t* l) {
  int4 z = {0, 0, 0, 0};
  *(int4*)l = z;
}

static __device__ __forceinline__ v8f wmma_bf16(v16b a, v16b b, v8f c) {
  return __builtin_amdgcn_wmma_f32_16x16x32_bf16(false, a, false, b, (short)0, c,
                                                 false, false);
}

// ---- WMMA fragment loaders (per CDNA5 ISA 7.12.2 16-bit layouts) ----
// A fragment 16x32 (MxK): lane l holds row (l&15); K chunks at (l>>4)*8 and +16.
static __device__ __forceinline__ v16b load_a_bf(const bf16_t* A, int lda) {
  int l = threadIdx.x & 31;
  const bf16_t* p = A + (l & 15) * lda + ((l >> 4) << 3);
  v16b r;
#pragma unroll
  for (int i = 0; i < 8; ++i) { r[i] = p[i]; r[8 + i] = p[16 + i]; }
  return r;
}
static __device__ __forceinline__ v16b load_a_f32(const float* A, int lda) {
  int l = threadIdx.x & 31;
  const float* p = A + (l & 15) * lda + ((l >> 4) << 3);
  v16b r;
#pragma unroll
  for (int i = 0; i < 8; ++i) { r[i] = tobf(p[i]); r[8 + i] = tobf(p[16 + i]); }
  return r;
}
// B fragment 32x16 (KxN) from "weight layout" W[n][k] (k contiguous):
// lane l = column n (l&15); K chunks at (l>>4)*8 and +16.
static __device__ __forceinline__ v16b load_b_bf(const bf16_t* W, int ldw) {
  int l = threadIdx.x & 31;
  const bf16_t* p = W + (l & 15) * ldw + ((l >> 4) << 3);
  v16b r;
#pragma unroll
  for (int i = 0; i < 8; ++i) { r[i] = p[i]; r[8 + i] = p[16 + i]; }
  return r;
}

// =====================  generic bf16 WMMA GEMM  =====================
// out[M,N] = A[M,K](bf16) @ W[N,K]^T(bf16) + bias, with fused epilogues.
// 128x128 block tile, 8 waves as 2x4, 4x2 fragments per wave, double-buffered LDS
// staged with async global->LDS copies (one barrier per K step).
// mode: 0=f32 store, 1=bf16 store, 2=bf16 gelu store,
//       3=outF += gate[b][col]*v, 4=outF += v
#define BM 128
#define BN 128
#define BK 32
#define LDSK (BK + 8)

__global__ __launch_bounds__(256)
void gemm_bf16(const bf16_t* __restrict__ A, const bf16_t* __restrict__ W,
               const float* __restrict__ bias,
               float* outF, bf16_t* outB,
               const float* __restrict__ gate, int gateStride, int rowsPerBatch,
               int M, int Nn, int K, int mode)
{
  __shared__ bf16_t As[2][BM][LDSK];
  __shared__ bf16_t Ws[2][BN][LDSK];
  int tid = threadIdx.x;
  int wid = tid >> 5;
  int wy = wid >> 2;            // 0..1  (64-row strip)
  int wx = wid & 3;             // 0..3  (32-col strip)
  int m0 = blockIdx.y * BM;
  int n0 = blockIdx.x * BN;

  int srow = tid >> 1;          // 0..127
  int sseg = (tid & 1) << 4;    // 0 or 16 (bf16 elements)

  auto stage = [&](int k0, int bufi) {
    int gm = m0 + srow;
    bf16_t* ad = &As[bufi][srow][sseg];
    if (gm < M) {
      const bf16_t* as = A + (size_t)gm * K + k0 + sseg;
      cp16(as, ad); cp16(as + 8, ad + 8);
    } else {
      zero16(ad); zero16(ad + 8);
    }
    const bf16_t* wsrc = W + (size_t)(n0 + srow) * K + k0 + sseg;
    bf16_t* wd = &Ws[bufi][srow][sseg];
    cp16(wsrc, wd); cp16(wsrc + 8, wd + 8);
  };

  v8f zero = {0.f,0.f,0.f,0.f,0.f,0.f,0.f,0.f};
  v8f acc[4][2];
#pragma unroll
  for (int i = 0; i < 4; ++i) { acc[i][0] = zero; acc[i][1] = zero; }

  int steps = K / BK;
  stage(0, 0);
  async_wait();
  __syncthreads();
  for (int kk = 0; kk < steps; ++kk) {
    int cur = kk & 1;
    if (kk + 1 < steps) stage((kk + 1) * BK, cur ^ 1);
    v16b b0 = load_b_bf(&Ws[cur][wx * 32][0],      LDSK);
    v16b b1 = load_b_bf(&Ws[cur][wx * 32 + 16][0], LDSK);
#pragma unroll
    for (int i = 0; i < 4; ++i) {
      v16b a = load_a_bf(&As[cur][wy * 64 + i * 16][0], LDSK);
      acc[i][0] = wmma_bf16(a, b0, acc[i][0]);
      acc[i][1] = wmma_bf16(a, b1, acc[i][1]);
    }
    async_wait();
    __syncthreads();
  }

  int l  = tid & 31;
  int cn = l & 15;
  int rb = (l >> 4) << 3;
#pragma unroll
  for (int i = 0; i < 4; ++i)
#pragma unroll
    for (int j = 0; j < 2; ++j) {
      int col = n0 + wx * 32 + j * 16 + cn;
#pragma unroll
      for (int rr = 0; rr < 8; ++rr) {
        int row = m0 + wy * 64 + i * 16 + rb + rr;
        if (row < M) {
          float v = acc[i][j][rr] + bias[col];
          size_t idx = (size_t)row * Nn + col;
          if (mode == 0) {
            outF[idx] = v;
          } else if (mode == 1) {
            outB[idx] = tobf(v);
          } else if (mode == 2) {
            float u = 0.7978845608028654f * (v + 0.044715f * v * v * v);
            outB[idx] = tobf(0.5f * v * (1.f + tanhf(u)));
          } else if (mode == 3) {
            int bb = row / rowsPerBatch;
            outF[idx] += gate[(size_t)bb * gateStride + col] * v;
          } else {
            outF[idx] += v;
          }
        }
      }
    }
}

// =====================  small elementwise kernels  =====================
__global__ void ss_kernel(const float* __restrict__ t, const float* __restrict__ tab,
                          float* __restrict__ ss) {
  int i = blockIdx.x * 256 + threadIdx.x;
  if (i < B_ * 6 * C_) ss[i] = t[i] + tab[i % (6 * C_)];
}

__global__ void cvt_kernel(const float* __restrict__ s, bf16_t* __restrict__ d, long n) {
  long i = (long)blockIdx.x * 256 + threadIdx.x;
  long st = (long)gridDim.x * 256;
  for (; i < n; i += st) d[i] = tobf(s[i]);
}

// LayerNorm + adaLN modulation -> bf16.  One wave per row (wave32).
__global__ __launch_bounds__(256)
void modln_kernel(const float* __restrict__ x, const float* __restrict__ ss,
                  bf16_t* __restrict__ out, int shIdx, int scIdx) {
  int wid = threadIdx.x >> 5, l = threadIdx.x & 31;
  int row = blockIdx.x * 8 + wid;
  const float* xr = x + (size_t)row * C_;
  float s = 0.f, s2 = 0.f;
  for (int c = l; c < C_; c += 32) { float v = xr[c]; s += v; s2 += v * v; }
#pragma unroll
  for (int o = 16; o > 0; o >>= 1) { s += __shfl_xor(s, o, 32); s2 += __shfl_xor(s2, o, 32); }
  float mean = s / C_;
  float var  = s2 / C_ - mean * mean;
  float inv  = rsqrtf(var + 1e-6f);
  int b = row / N_;
  const float* sh = ss + (size_t)b * 6 * C_ + (size_t)shIdx * C_;
  const float* sc = ss + (size_t)b * 6 * C_ + (size_t)scIdx * C_;
  bf16_t* orow = out + (size_t)row * C_;
  for (int c = l; c < C_; c += 32) {
    float v = (xr[c] - mean) * inv;
    orow[c] = tobf(v * (1.f + sc[c]) + sh[c]);
  }
}

// stage one 96-wide padded head row: 72 bf16 from global (async) + 24 zeros
static __device__ __forceinline__ void stage_row96(const bf16_t* g, bf16_t* l) {
#pragma unroll
  for (int i = 0; i < 9; ++i) cp16(g + i * 8, l + i * 8);
  zero16(l + 72); zero16(l + 80); zero16(l + 88);
}
static __device__ __forceinline__ void zero_row(bf16_t* l, int n8) {
  for (int i = 0; i < n8; ++i) zero16(l + i * 8);
}

// =====================  spatial attention (len 256)  =====================
// grid.x = B*T*NH; dyn LDS: Kb 256x96 bf16 | Vt 80x256 bf16 | Qb 64x96 bf16 | Sf 64x256 f32
__global__ __launch_bounds__(256)
void attn_spatial(const bf16_t* __restrict__ qkv, bf16_t* __restrict__ outO) {
  extern __shared__ char smem[];
  bf16_t* Kb = (bf16_t*)smem;
  bf16_t* Vt = Kb + 256 * 96;
  bf16_t* Qb = Vt + 80 * 256;
  float*  Sf = (float*)(Qb + 64 * 96);
  int bt = blockIdx.x / NH_;
  int h  = blockIdx.x % NH_;
  int tid = threadIdx.x, wid = tid >> 5, l = tid & 31;
  {
    int n = tid;
    const bf16_t* krow = qkv + (size_t)(bt * S_ + n) * C3_ + (NH_ + h) * HD_;
    const bf16_t* vrow = qkv + (size_t)(bt * S_ + n) * C3_ + (2 * NH_ + h) * HD_;
    stage_row96(krow, Kb + n * 96);
    for (int d = 0; d < 80; ++d) Vt[d * 256 + n] = (d < HD_) ? fromb(vrow[d]) : tobf(0.f);
  }
  async_wait();
  __syncthreads();
  for (int qt = 0; qt < 4; ++qt) {
    {   // stage Q 64x96: each thread a 24-elem (48B) chunk
      int r = tid >> 2, cg = (tid & 3) * 24;
      bf16_t* dst = Qb + r * 96 + cg;
      if (cg < HD_) {
        const bf16_t* qrow = qkv + (size_t)(bt * S_ + qt * 64 + r) * C3_ + h * HD_ + cg;
        cp16(qrow, dst); cp16(qrow + 8, dst + 8); cp16(qrow + 16, dst + 16);
      } else {
        zero_row(dst, 3);
      }
    }
    async_wait();
    __syncthreads();
    {   // scores 64x256 via WMMA: wave (wy 0..3) rows, (wxn 0..1) 128-col half
      int wy = wid >> 1, wxn = wid & 1;
      v8f zero = {0.f,0.f,0.f,0.f,0.f,0.f,0.f,0.f};
      v8f acc[8];
#pragma unroll
      for (int j = 0; j < 8; ++j) acc[j] = zero;
      for (int k0 = 0; k0 < 96; k0 += 32) {
        v16b a = load_a_bf(Qb + (wy * 16) * 96 + k0, 96);
#pragma unroll
        for (int j = 0; j < 8; ++j) {
          v16b bb = load_b_bf(Kb + (size_t)(wxn * 128 + j * 16) * 96 + k0, 96);
          acc[j] = wmma_bf16(a, bb, acc[j]);
        }
      }
      int rb = (l >> 4) << 3;
#pragma unroll
      for (int j = 0; j < 8; ++j) {
        int n = wxn * 128 + j * 16 + (l & 15);
#pragma unroll
        for (int rr = 0; rr < 8; ++rr) {
          int m = wy * 16 + rb + rr;
          Sf[m * 256 + n] = acc[j][rr] * SCALE_;
        }
      }
    }
    __syncthreads();
    // softmax: wave handles 8 rows, lanes stride cols
    for (int r = 0; r < 8; ++r) {
      int m = wid * 8 + r;
      float* srow = Sf + m * 256;
      float v[8]; float mx = -1e30f;
#pragma unroll
      for (int i = 0; i < 8; ++i) { v[i] = srow[l + i * 32]; mx = fmaxf(mx, v[i]); }
#pragma unroll
      for (int o = 16; o > 0; o >>= 1) mx = fmaxf(mx, __shfl_xor(mx, o, 32));
      float sum = 0.f;
#pragma unroll
      for (int i = 0; i < 8; ++i) { v[i] = __expf(v[i] - mx); sum += v[i]; }
#pragma unroll
      for (int o = 16; o > 0; o >>= 1) sum += __shfl_xor(sum, o, 32);
      float is = 1.f / sum;
#pragma unroll
      for (int i = 0; i < 8; ++i) srow[l + i * 32] = v[i] * is;
    }
    __syncthreads();
    {   // O = S @ V : out 64x72 (padded 80)
      int wy = wid >> 1, wxn = wid & 1;
      int j0 = wxn ? 3 : 0, cnt = wxn ? 2 : 3;
      v8f zero = {0.f,0.f,0.f,0.f,0.f,0.f,0.f,0.f};
      v8f acc[3]; acc[0] = zero; acc[1] = zero; acc[2] = zero;
      for (int k0 = 0; k0 < 256; k0 += 32) {
        v16b a = load_a_f32(Sf + (wy * 16) * 256 + k0, 256);
        for (int jj = 0; jj < cnt; ++jj) {
          v16b bb = load_b_bf(Vt + (size_t)((j0 + jj) * 16) * 256 + k0, 256);
          acc[jj] = wmma_bf16(a, bb, acc[jj]);
        }
      }
      int rb = (l >> 4) << 3;
      for (int jj = 0; jj < cnt; ++jj) {
        int d = (j0 + jj) * 16 + (l & 15);
        if (d < HD_) {
#pragma unroll
          for (int rr = 0; rr < 8; ++rr) {
            int m = wy * 16 + rb + rr;
            outO[(size_t)(bt * S_ + qt * 64 + m) * C_ + h * HD_ + d] = tobf(acc[jj][rr]);
          }
        }
      }
    }
    __syncthreads();
  }
}

// =====================  temporal attention (len 32, causal + RoPE)  ============
// grid.x = B*S*2; each wave owns one head (8 heads per block).
#define TSLOT_B 21504
__global__ __launch_bounds__(256)
void attn_temporal(const bf16_t* __restrict__ qkv, const float* __restrict__ rope,
                   bf16_t* __restrict__ outO) {
  extern __shared__ char smem[];
  int tid = threadIdx.x, wid = tid >> 5, l = tid & 31;
  char* basep = smem + (size_t)wid * TSLOT_B;
  bf16_t* Qb = (bf16_t*)basep;          // 32x96
  bf16_t* Kb = Qb + 32 * 96;            // 32x96
  bf16_t* Vt = Kb + 32 * 96;            // 80x32
  float*  Sf = (float*)(Vt + 80 * 32);  // 32x32
  int bs = blockIdx.x >> 1, hh = blockIdx.x & 1;
  int b = bs / S_, s = bs % S_;
  int h = hh * 8 + wid;
  {   // stage with RoPE; lane l owns time-row m = l
    int m = l;
    size_t row = (size_t)b * N_ + (size_t)m * S_ + s;
    const bf16_t* qrow = qkv + row * C3_ + h * HD_;
    const bf16_t* krow = qkv + row * C3_ + (NH_ + h) * HD_;
    const bf16_t* vrow = qkv + row * C3_ + (2 * NH_ + h) * HD_;
    for (int i = 0; i < HD2_; ++i) {
      float ang = rope[m * HD2_ + i];
      float cs = __cosf(ang), sn = __sinf(ang);
      float qe = fromb(qrow[2 * i]), qo = fromb(qrow[2 * i + 1]);
      Qb[m * 96 + 2 * i]     = tobf(qe * cs - qo * sn);
      Qb[m * 96 + 2 * i + 1] = tobf(qe * sn + qo * cs);
      float ke = fromb(krow[2 * i]), ko = fromb(krow[2 * i + 1]);
      Kb[m * 96 + 2 * i]     = tobf(ke * cs - ko * sn);
      Kb[m * 96 + 2 * i + 1] = tobf(ke * sn + ko * cs);
    }
    for (int k = HD_; k < 96; ++k) { Qb[m * 96 + k] = tobf(0.f); Kb[m * 96 + k] = tobf(0.f); }
    for (int d = 0; d < 80; ++d) Vt[d * 32 + m] = (d < HD_) ? fromb(vrow[d]) : tobf(0.f);
  }
  __syncthreads();
  {   // scores 32x32 + causal mask
    v8f zero = {0.f,0.f,0.f,0.f,0.f,0.f,0.f,0.f};
    v8f acc[2][2];
    acc[0][0] = zero; acc[0][1] = zero; acc[1][0] = zero; acc[1][1] = zero;
    for (int k0 = 0; k0 < 96; k0 += 32) {
      v16b a0 = load_a_bf(Qb + k0, 96);
      v16b a1 = load_a_bf(Qb + 16 * 96 + k0, 96);
      v16b b0 = load_b_bf(Kb + k0, 96);
      v16b b1 = load_b_bf(Kb + 16 * 96 + k0, 96);
      acc[0][0] = wmma_bf16(a0, b0, acc[0][0]);
      acc[0][1] = wmma_bf16(a0, b1, acc[0][1]);
      acc[1][0] = wmma_bf16(a1, b0, acc[1][0]);
      acc[1][1] = wmma_bf16(a1, b1, acc[1][1]);
    }
    int rb = (l >> 4) << 3;
#pragma unroll
    for (int i = 0; i < 2; ++i)
#pragma unroll
      for (int j = 0; j < 2; ++j) {
        int n = j * 16 + (l & 15);
#pragma unroll
        for (int rr = 0; rr < 8; ++rr) {
          int m = i * 16 + rb + rr;
          Sf[m * 32 + n] = (n <= m) ? acc[i][j][rr] * SCALE_ : -1e9f;
        }
      }
  }
  __syncthreads();
  {   // softmax: lane l owns row l, serial over 32 cols
    float* srow = Sf + l * 32;
    float mx = -1e30f;
    for (int c = 0; c < 32; ++c) mx = fmaxf(mx, srow[c]);
    float sum = 0.f;
    for (int c = 0; c < 32; ++c) { float e = __expf(srow[c] - mx); srow[c] = e; sum += e; }
    float is = 1.f / sum;
    for (int c = 0; c < 32; ++c) srow[c] *= is;
  }
  __syncthreads();
  {   // O = S @ V (K = 32, single WMMA step)
    v8f zero = {0.f,0.f,0.f,0.f,0.f,0.f,0.f,0.f};
    v16b a0 = load_a_f32(Sf, 32);
    v16b a1 = load_a_f32(Sf + 16 * 32, 32);
    int rb = (l >> 4) << 3;
#pragma unroll
    for (int j = 0; j < 5; ++j) {
      v16b bb = load_b_bf(Vt + j * 16 * 32, 32);
      v8f c0 = wmma_bf16(a0, bb, zero);
      v8f c1 = wmma_bf16(a1, bb, zero);
      int d = j * 16 + (l & 15);
      if (d < HD_) {
#pragma unroll
        for (int rr = 0; rr < 8; ++rr) {
          int m0r = rb + rr, m1r = 16 + rb + rr;
          outO[((size_t)b * N_ + (size_t)m0r * S_ + s) * C_ + h * HD_ + d] = tobf(c0[rr]);
          outO[((size_t)b * N_ + (size_t)m1r * S_ + s) * C_ + h * HD_ + d] = tobf(c1[rr]);
        }
      }
    }
  }
}

// =====================  cross attention (kv len 120 -> padded 128)  ============
// grid.x = B*NH*(N/64); dyn LDS: Kb 128x96 | Vt 80x128 | Qb 64x96 | Sf 64x128 f32
__global__ __launch_bounds__(256)
void attn_cross(const bf16_t* __restrict__ Qc, const bf16_t* __restrict__ kvc,
                bf16_t* __restrict__ outO) {
  extern __shared__ char smem[];
  bf16_t* Kb = (bf16_t*)smem;
  bf16_t* Vt = Kb + 128 * 96;
  bf16_t* Qb = Vt + 80 * 128;
  float*  Sf = (float*)(Qb + 64 * 96);
  int qt = blockIdx.x & 127;
  int bh = blockIdx.x >> 7;
  int b = bh / NH_, h = bh % NH_;
  int tid = threadIdx.x, wid = tid >> 5, l = tid & 31;
  {
    int n = tid & 127, half = tid >> 7;
    const bf16_t* base = kvc + (size_t)(b * LY_ + n) * (2 * C_);
    bf16_t* kd = Kb + n * 96 + half * 48;
    if (n < LY_) {
      if (half == 0) {   // k = 0..47
        const bf16_t* g = base + h * HD_;
        cp16(g, kd); cp16(g + 8, kd + 8); cp16(g + 16, kd + 16);
        cp16(g + 24, kd + 24); cp16(g + 32, kd + 32); cp16(g + 40, kd + 40);
      } else {           // k = 48..71 valid, 72..95 zero
        const bf16_t* g = base + h * HD_ + 48;
        cp16(g, kd); cp16(g + 8, kd + 8); cp16(g + 16, kd + 16);
        zero_row(kd + 24, 3);
      }
    } else {
      zero_row(kd, 6);
    }
    for (int i = 0; i < 40; ++i) {
      int d = half * 40 + i;
      Vt[d * 128 + n] = (n < LY_ && d < HD_) ? fromb(base[(NH_ + h) * HD_ + d]) : tobf(0.f);
    }
  }
  {   // stage Q 64x96: each thread a 24-elem chunk
    int r = tid >> 2, cg = (tid & 3) * 24;
    bf16_t* dst = Qb + r * 96 + cg;
    if (cg < HD_) {
      const bf16_t* qrow = Qc + ((size_t)b * N_ + qt * 64 + r) * C_ + h * HD_ + cg;
      cp16(qrow, dst); cp16(qrow + 8, dst + 8); cp16(qrow + 16, dst + 16);
    } else {
      zero_row(dst, 3);
    }
  }
  async_wait();
  __syncthreads();
  {   // scores 64x128
    int wy = wid >> 1, wxn = wid & 1;
    v8f zero = {0.f,0.f,0.f,0.f,0.f,0.f,0.f,0.f};
    v8f acc[4];
#pragma unroll
    for (int j = 0; j < 4; ++j) acc[j] = zero;
    for (int k0 = 0; k0 < 96; k0 += 32) {
      v16b a = load_a_bf(Qb + (wy * 16) * 96 + k0, 96);
#pragma unroll
      for (int j = 0; j < 4; ++j) {
        v16b bb = load_b_bf(Kb + (size_t)(wxn * 64 + j * 16) * 96 + k0, 96);
        acc[j] = wmma_bf16(a, bb, acc[j]);
      }
    }
    int rb = (l >> 4) << 3;
#pragma unroll
    for (int j = 0; j < 4; ++j) {
      int n = wxn * 64 + j * 16 + (l & 15);
#pragma unroll
      for (int rr = 0; rr < 8; ++rr) {
        int m = wy * 16 + rb + rr;
        Sf[m * 128 + n] = (n < LY_) ? acc[j][rr] * SCALE_ : -1e9f;
      }
    }
  }
  __syncthreads();
  for (int r = 0; r < 8; ++r) {   // softmax over 128 cols
    int m = wid * 8 + r;
    float* srow = Sf + m * 128;
    float v[4]; float mx = -1e30f;
#pragma unroll
    for (int i = 0; i < 4; ++i) { v[i] = srow[l + i * 32]; mx = fmaxf(mx, v[i]); }
#pragma unroll
    for (int o = 16; o > 0; o >>= 1) mx = fmaxf(mx, __shfl_xor(mx, o, 32));
    float sum = 0.f;
#pragma unroll
    for (int i = 0; i < 4; ++i) { v[i] = __expf(v[i] - mx); sum += v[i]; }
#pragma unroll
    for (int o = 16; o > 0; o >>= 1) sum += __shfl_xor(sum, o, 32);
    float is = 1.f / sum;
#pragma unroll
    for (int i = 0; i < 4; ++i) srow[l + i * 32] = v[i] * is;
  }
  __syncthreads();
  {   // O = S @ V  (K = 128)
    int wy = wid >> 1, wxn = wid & 1;
    int j0 = wxn ? 3 : 0, cnt = wxn ? 2 : 3;
    v8f zero = {0.f,0.f,0.f,0.f,0.f,0.f,0.f,0.f};
    v8f acc[3]; acc[0] = zero; acc[1] = zero; acc[2] = zero;
    for (int k0 = 0; k0 < 128; k0 += 32) {
      v16b a = load_a_f32(Sf + (wy * 16) * 128 + k0, 128);
      for (int jj = 0; jj < cnt; ++jj) {
        v16b bb = load_b_bf(Vt + (size_t)((j0 + jj) * 16) * 128 + k0, 128);
        acc[jj] = wmma_bf16(a, bb, acc[jj]);
      }
    }
    int rb = (l >> 4) << 3;
    for (int jj = 0; jj < cnt; ++jj) {
      int d = (j0 + jj) * 16 + (l & 15);
      if (d < HD_) {
#pragma unroll
        for (int rr = 0; rr < 8; ++rr) {
          int m = wy * 16 + rb + rr;
          outO[((size_t)b * N_ + qt * 64 + m) * C_ + h * HD_ + d] = tobf(acc[jj][rr]);
        }
      }
    }
  }
}

// ============================  host orchestration  ============================
static void cvt(const float* s, bf16_t* d, size_t n, hipStream_t st) {
  int blocks = (int)((n + 2047) / 2048);
  cvt_kernel<<<blocks, 256, 0, st>>>(s, d, (long)n);
}

extern "C" void kernel_launch(void* const* d_in, const int* in_sizes, int n_in,
                              void* d_out, int out_size, void* d_ws, size_t ws_size,
                              hipStream_t stream) {
  const float* x    = (const float*)d_in[0];
  const float* y    = (const float*)d_in[1];
  const float* t    = (const float*)d_in[2];
  const float* sst  = (const float*)d_in[3];
  const float* aqkv_w = (const float*)d_in[4];
  const float* aqkv_b = (const float*)d_in[5];
  const float* aprj_w = (const float*)d_in[6];
  const float* aprj_b = (const float*)d_in[7];
  const float* tqkv_w = (const float*)d_in[8];
  const float* tqkv_b = (const float*)d_in[9];
  const float* tprj_w = (const float*)d_in[10];
  const float* tprj_b = (const float*)d_in[11];
  const float* cq_w   = (const float*)d_in[12];
  const float* cq_b   = (const float*)d_in[13];
  const float* ckv_w  = (const float*)d_in[14];
  const float* ckv_b  = (const float*)d_in[15];
  const float* cprj_w = (const float*)d_in[16];
  const float* cprj_b = (const float*)d_in[17];
  const float* fc1_w  = (const float*)d_in[18];
  const float* fc1_b  = (const float*)d_in[19];
  const float* fc2_w  = (const float*)d_in[20];
  const float* fc2_b  = (const float*)d_in[21];
  const float* rope   = (const float*)d_in[22];
  float* xr = (float*)d_out;

  char* w = (char*)d_ws;
  auto take = [&](size_t bytes) -> char* {
    char* p = w; w += (bytes + 255) & ~(size_t)255; return p;
  };
  float*  ss      = (float*)take((size_t)B_ * 6 * C_ * 4);
  bf16_t* wqkv_s  = (bf16_t*)take((size_t)C3_ * C_ * 2);
  bf16_t* wproj_s = (bf16_t*)take((size_t)C_ * C_ * 2);
  bf16_t* wqkv_t  = (bf16_t*)take((size_t)C3_ * C_ * 2);
  bf16_t* wproj_t = (bf16_t*)take((size_t)C_ * C_ * 2);
  bf16_t* wq_c    = (bf16_t*)take((size_t)C_ * C_ * 2);
  bf16_t* wkv_c   = (bf16_t*)take((size_t)2 * C_ * C_ * 2);
  bf16_t* wproj_c = (bf16_t*)take((size_t)C_ * C_ * 2);
  bf16_t* wfc1    = (bf16_t*)take((size_t)MLPH_ * C_ * 2);
  bf16_t* wfc2    = (bf16_t*)take((size_t)C_ * MLPH_ * 2);
  bf16_t* ybf     = (bf16_t*)take((size_t)B_ * LY_ * C_ * 2);
  bf16_t* kvc     = (bf16_t*)take((size_t)B_ * LY_ * 2 * C_ * 2);
  bf16_t* xm      = (bf16_t*)take((size_t)M_ * C_ * 2);
  bf16_t* xb      = (bf16_t*)take((size_t)M_ * C_ * 2);
  bf16_t* obuf    = (bf16_t*)take((size_t)M_ * C_ * 2);
  bf16_t* big     = (bf16_t*)take((size_t)M_ * MLPH_ * 2);

  dim3 blk(256);
  // adaLN shift/scale/gate table
  ss_kernel<<<(B_ * 6 * C_ + 255) / 256, blk, 0, stream>>>(t, sst, ss);
  // bf16 weight copies (all fit in the 192MB L2)
  cvt(aqkv_w, wqkv_s, (size_t)C3_ * C_, stream);
  cvt(aprj_w, wproj_s, (size_t)C_ * C_, stream);
  cvt(tqkv_w, wqkv_t, (size_t)C3_ * C_, stream);
  cvt(tprj_w, wproj_t, (size_t)C_ * C_, stream);
  cvt(cq_w,   wq_c,   (size_t)C_ * C_, stream);
  cvt(ckv_w,  wkv_c,  (size_t)2 * C_ * C_, stream);
  cvt(cprj_w, wproj_c,(size_t)C_ * C_, stream);
  cvt(fc1_w,  wfc1,   (size_t)MLPH_ * C_, stream);
  cvt(fc2_w,  wfc2,   (size_t)C_ * MLPH_, stream);
  cvt(y,      ybf,    (size_t)B_ * LY_ * C_, stream);
  // running residual x lives in d_out
  (void)hipMemcpyAsync(xr, x, (size_t)M_ * C_ * 4, hipMemcpyDeviceToDevice, stream);

  // ---- spatial attention branch ----
  modln_kernel<<<M_ / 8, blk, 0, stream>>>(xr, ss, xm, 0, 1);
  gemm_bf16<<<dim3(C3_ / BN, M_ / BM), blk, 0, stream>>>(
      xm, wqkv_s, aqkv_b, nullptr, big, nullptr, 0, N_, M_, C3_, C_, 1);
  attn_spatial<<<B_ * T_ * NH_, blk, 167936, stream>>>(big, obuf);
  gemm_bf16<<<dim3(C_ / BN, M_ / BM), blk, 0, stream>>>(
      obuf, wproj_s, aprj_b, xr, nullptr, ss + 2 * C_, 6 * C_, N_, M_, C_, C_, 3);

  // ---- temporal attention branch (causal + RoPE) ----
  cvt(xr, xb, (size_t)M_ * C_, stream);
  gemm_bf16<<<dim3(C3_ / BN, M_ / BM), blk, 0, stream>>>(
      xb, wqkv_t, tqkv_b, nullptr, big, nullptr, 0, N_, M_, C3_, C_, 1);
  attn_temporal<<<B_ * S_ * 2, blk, 172032, stream>>>(big, rope, obuf);
  gemm_bf16<<<dim3(C_ / BN, M_ / BM), blk, 0, stream>>>(
      obuf, wproj_t, tprj_b, xr, nullptr, ss + 2 * C_, 6 * C_, N_, M_, C_, C_, 3);

  // ---- cross attention ----
  cvt(xr, xb, (size_t)M_ * C_, stream);
  gemm_bf16<<<dim3(C_ / BN, M_ / BM), blk, 0, stream>>>(
      xb, wq_c, cq_b, nullptr, obuf, nullptr, 0, N_, M_, C_, C_, 1);
  gemm_bf16<<<dim3((2 * C_) / BN, (B_ * LY_ + BM - 1) / BM), blk, 0, stream>>>(
      ybf, wkv_c, ckv_b, nullptr, kvc, nullptr, 0, N_, B_ * LY_, 2 * C_, C_, 1);
  attn_cross<<<B_ * NH_ * (N_ / 64), blk, 90112, stream>>>(obuf, kvc, xm);
  gemm_bf16<<<dim3(C_ / BN, M_ / BM), blk, 0, stream>>>(
      xm, wproj_c, cprj_b, xr, nullptr, nullptr, 0, N_, M_, C_, C_, 4);

  // ---- MLP ----
  modln_kernel<<<M_ / 8, blk, 0, stream>>>(xr, ss, xm, 3, 4);
  gemm_bf16<<<dim3(MLPH_ / BN, M_ / BM), blk, 0, stream>>>(
      xm, wfc1, fc1_b, nullptr, big, nullptr, 0, N_, M_, MLPH_, C_, 2);
  gemm_bf16<<<dim3(C_ / BN, M_ / BM), blk, 0, stream>>>(
      big, wfc2, fc2_b, xr, nullptr, ss + 5 * C_, 6 * C_, N_, M_, C_, MLPH_, 3);
}
